// MambaUnet_88467736363294
// MI455X (gfx1250) — compile-verified, matrix operand order checked by reading, not run on hardware
//
#include <hip/hip_runtime.h>
#include <math.h>

// ---------------------------------------------------------------- types
typedef __bf16 bhalf_t;
typedef __attribute__((ext_vector_type(4)))  __bf16 v4bf;
typedef __attribute__((ext_vector_type(8)))  __bf16 v8bf;
typedef __attribute__((ext_vector_type(16))) __bf16 v16bf;
typedef __attribute__((ext_vector_type(8)))  float  v8f;

#define BB 16
#define SS 5
#define HH 64
#define WW 64
#define DD 128
#define DSS 16
#define LL (HH*WW)
#define BL (BB*LL)

// ---------------------------------------------------------------- helpers
__device__ __forceinline__ float apply_act(float v, int act) {
  switch (act) {
    case 1: return v > 0.f ? v : 0.f;                                   // relu
    case 2: return 0.5f * v * (1.f + erff(v * 0.7071067811865475f));    // exact gelu
    case 3: return v / (1.f + __expf(-v));                              // silu
    case 4: return 1.f / (1.f + __expf(-v));                            // sigmoid
  }
  return v;
}

// 4-wide loads (128-bit for fp32, 64-bit for bf16)
__device__ __forceinline__ void ld4(const float* p, float t[4]) {
  float4 v = *(const float4*)p;
  t[0] = v.x; t[1] = v.y; t[2] = v.z; t[3] = v.w;
}
__device__ __forceinline__ void ld4(const bhalf_t* p, float t[4]) {
  v4bf v = *(const v4bf*)p;
  t[0] = (float)v[0]; t[1] = (float)v[1]; t[2] = (float)v[2]; t[3] = (float)v[3];
}

// ---------------------------------------------------------------- GEMM (WMMA bf16)
// C[M,N] = act(A[M,K] @ Wt[N,K]^T + bias) (* mul) (+ res), stored through an
// optional row-index map (ConvT parity scatter) with ldc/col_off (concat).
// Assumes M % 64 == 0 (true for every launch in this pipeline).
// Tile: 64x64 C per block, K stepped by 64 (2 WMMA K-chunks per stage).
struct GemmArgs {
  const float* Wt;     // [N,K] row-major (torch weight layout for x @ W.T)
  const float* bias;   // [N] or nullptr
  float*       out;
  const float* res;    // residual read at out index, or nullptr
  const float* mul;    // elementwise multiplier read at out index, or nullptr
  int M, N, K, lda, ldc, col_off, act;
  int map_mode, rpi, wc, img_str, rsy, rsx, roff;
};

template <typename TA>
__global__ __launch_bounds__(256) void gemm_wmma(const TA* __restrict__ A, GemmArgs g) {
  __shared__ bhalf_t As[64][72];   // 144B rows: v4bf stores + b128 fragment reads aligned
  __shared__ bhalf_t Bs[64][72];
  const int tid  = threadIdx.x;
  const int lane = tid & 31;
  const int wave = tid >> 5;
  const int m0 = blockIdx.y * 64;
  const int n0 = blockIdx.x * 64;
  const int wm = (wave & 3) * 16;     // 4 row groups of 16
  const int wn = (wave >> 2) * 32;    // 2 col groups of 32
  const int hf  = lane >> 4;
  const int l16 = lane & 15;

  v8f acc0 = {0.f,0.f,0.f,0.f,0.f,0.f,0.f,0.f};
  v8f acc1 = {0.f,0.f,0.f,0.f,0.f,0.f,0.f,0.f};

  const int srow = tid >> 2;          // staging row (0..63)
  const int scol = (tid & 3) * 4;     // staging col base; +{0,16,32,48}

  for (int k0 = 0; k0 < g.K; k0 += 64) {
    const bool kfull = (k0 + 64 <= g.K);
    const bool nfull = (n0 + 64 <= g.N);
    // ---- stage A (64x64): batch 4 independent wide loads, then convert+store
    if (kfull) {
      const TA* arow = A + (long)(m0 + srow) * g.lda + (k0 + scol);
      float t[4][4];
#pragma unroll
      for (int it = 0; it < 4; ++it) ld4(arow + it * 16, t[it]);
      if (k0 + 64 < g.K) __builtin_prefetch((const void*)(arow + 64), 0, 1);
#pragma unroll
      for (int it = 0; it < 4; ++it) {
        v4bf s = {(bhalf_t)t[it][0], (bhalf_t)t[it][1], (bhalf_t)t[it][2], (bhalf_t)t[it][3]};
        *(v4bf*)&As[srow][scol + it * 16] = s;
      }
    } else {
      for (int i = tid; i < 64 * 64; i += 256) {
        int m = i >> 6, k = i & 63;
        int gk = k0 + k;
        float v = 0.f;
        if (gk < g.K) v = (float)A[(long)(m0 + m) * g.lda + gk];
        As[m][k] = (bhalf_t)v;
      }
    }
    // ---- stage B (64x64) from weights [N,K]
    if (kfull && nfull) {
      const float* brow = g.Wt + (long)(n0 + srow) * g.K + (k0 + scol);
      float t[4][4];
#pragma unroll
      for (int it = 0; it < 4; ++it) ld4(brow + it * 16, t[it]);
      if (k0 + 64 < g.K) __builtin_prefetch((const void*)(brow + 64), 0, 1);
#pragma unroll
      for (int it = 0; it < 4; ++it) {
        v4bf s = {(bhalf_t)t[it][0], (bhalf_t)t[it][1], (bhalf_t)t[it][2], (bhalf_t)t[it][3]};
        *(v4bf*)&Bs[srow][scol + it * 16] = s;
      }
    } else {
      for (int i = tid; i < 64 * 64; i += 256) {
        int n = i >> 6, k = i & 63;
        int gn = n0 + n, gk = k0 + k;
        float v = 0.f;
        if (gn < g.N && gk < g.K) v = g.Wt[(long)gn * g.K + gk];
        Bs[n][k] = (bhalf_t)v;
      }
    }
    __syncthreads();

    // ---- fragments: all b128 LDS reads issued up front, then 4 WMMAs
#pragma unroll
    for (int kk = 0; kk < 2; ++kk) {
      const int kb = kk * 32;
      v8bf alo  = *(const v8bf*)&As[wm + l16][kb + hf * 8];
      v8bf ahi  = *(const v8bf*)&As[wm + l16][kb + 16 + hf * 8];
      v8bf b0lo = *(const v8bf*)&Bs[wn + l16][kb + hf * 8];
      v8bf b0hi = *(const v8bf*)&Bs[wn + l16][kb + 16 + hf * 8];
      v8bf b1lo = *(const v8bf*)&Bs[wn + 16 + l16][kb + hf * 8];
      v8bf b1hi = *(const v8bf*)&Bs[wn + 16 + l16][kb + 16 + hf * 8];
      v16bf af  = __builtin_shufflevector(alo,  ahi,  0,1,2,3,4,5,6,7,8,9,10,11,12,13,14,15);
      v16bf bf0 = __builtin_shufflevector(b0lo, b0hi, 0,1,2,3,4,5,6,7,8,9,10,11,12,13,14,15);
      v16bf bf1 = __builtin_shufflevector(b1lo, b1hi, 0,1,2,3,4,5,6,7,8,9,10,11,12,13,14,15);
      acc0 = __builtin_amdgcn_wmma_f32_16x16x32_bf16(false, af, false, bf0, (short)0, acc0, false, false);
      acc1 = __builtin_amdgcn_wmma_f32_16x16x32_bf16(false, af, false, bf1, (short)0, acc1, false, false);
    }
    __syncthreads();
  }

  // ---- epilogue
#pragma unroll
  for (int j = 0; j < 2; ++j) {
    v8f acc = j ? acc1 : acc0;
    int gn = n0 + wn + j * 16 + l16;
    if (gn >= g.N) continue;
#pragma unroll
    for (int r = 0; r < 8; ++r) {
      int gm = m0 + wm + r + hf * 8;
      float v = acc[r];
      if (g.bias) v += g.bias[gn];
      v = apply_act(v, g.act);
      long orow;
      if (g.map_mode == 0) {
        orow = gm;
      } else {
        int b  = gm / g.rpi;
        int rr = gm % g.rpi;
        orow = (long)b * g.img_str + (rr / g.wc) * (long)g.rsy + (rr % g.wc) * (long)g.rsx + g.roff;
      }
      long oidx = orow * g.ldc + g.col_off + gn;
      if (g.mul) v *= g.mul[oidx];
      if (g.res) v += g.res[oidx];
      g.out[oidx] = v;
    }
  }
}

// ---------------------------------------------------------------- embed
// emb[b,l,d] = sum_s wts[s] * (conv3x3(x[b,s])[l,d] + emb_b[d] + pos[l,d] + time[s,d])
__global__ __launch_bounds__(256) void embed_kernel(
    const float* __restrict__ x, const float* __restrict__ ew, const float* __restrict__ eb,
    const float* __restrict__ pos, const float* __restrict__ te, float* __restrict__ emb) {
  __shared__ float wx[9][WW];
  __shared__ float tvec[DD];
  const int b = blockIdx.x / HH;
  const int y = blockIdx.x % HH;
  const int tid = threadIdx.x;

  if (tid < DD) {
    float t = 0.f;
    for (int s = 0; s < SS; ++s) {
      float w = (s == SS - 1) ? 1.0f : 0.5f;
      t += w * te[s * DD + tid];
    }
    tvec[tid] = t;
  }
  for (int i = tid; i < 9 * WW; i += 256) {
    int j = i / WW, xx = i % WW;
    int dy = j / 3 - 1, dx = j % 3 - 1;
    int iy = y + dy, ix = xx + dx;
    float v = 0.f;
    if (iy >= 0 && iy < HH && ix >= 0 && ix < WW) {
      for (int s = 0; s < SS; ++s) {
        float w = (s == SS - 1) ? 1.0f : 0.5f;
        v += w * x[(((long)b * SS + s) * HH + iy) * WW + ix];
      }
    }
    wx[j][xx] = v;
  }
  __syncthreads();
  float wtsum = 0.5f * (SS - 1) + 1.0f;
  for (int o = tid; o < WW * DD; o += 256) {
    int xx = o / DD, d = o % DD;
    int l = y * WW + xx;
    float acc = tvec[d] + wtsum * (eb[d] + pos[(long)l * DD + d]);
#pragma unroll
    for (int j = 0; j < 9; ++j) acc += wx[j][xx] * ew[d * 9 + j];
    emb[(((long)b * LL) + l) * DD + d] = acc;
  }
}

// ---------------------------------------------------------------- layernorm (wave per row)
__global__ __launch_bounds__(256) void ln_kernel(const float* __restrict__ x,
                                                 const float* __restrict__ w,
                                                 const float* __restrict__ bia,
                                                 float* __restrict__ y, int rows) {
  int lane = threadIdx.x & 31;
  int wave = threadIdx.x >> 5;
  int row = blockIdx.x * 8 + wave;
  if (row >= rows) return;
  const float* xr = x + (long)row * DD;
  float s[4];
  float sum = 0.f;
#pragma unroll
  for (int j = 0; j < 4; ++j) { s[j] = xr[lane + 32 * j]; sum += s[j]; }
#pragma unroll
  for (int off = 16; off > 0; off >>= 1) sum += __shfl_xor(sum, off, 32);
  float mean = sum * (1.0f / DD);
  float vs = 0.f;
#pragma unroll
  for (int j = 0; j < 4; ++j) { float d = s[j] - mean; vs += d * d; }
#pragma unroll
  for (int off = 16; off > 0; off >>= 1) vs += __shfl_xor(vs, off, 32);
  float rstd = rsqrtf(vs * (1.0f / DD) + 1e-5f);
#pragma unroll
  for (int j = 0; j < 4; ++j) {
    int d = lane + 32 * j;
    y[(long)row * DD + d] = (s[j] - mean) * rstd * w[d] + bia[d];
  }
}

// ---------------------------------------------------------------- depthwise conv1d k=4, left pad 3, residual add
__global__ __launch_bounds__(256) void dwconv_kernel(const float* __restrict__ xn,
                                                     const float* __restrict__ w,
                                                     const float* __restrict__ bia,
                                                     float* __restrict__ emb) {
  long idx = (long)blockIdx.x * 256 + threadIdx.x;
  if (idx >= (long)BL * DD) return;
  int d = idx % DD;
  long bl = idx / DD;
  int l = bl % LL;
  long base = (bl - l) * DD;
  float acc = bia[d];
#pragma unroll
  for (int k = 0; k < 4; ++k) {
    int ls = l - 3 + k;
    if (ls >= 0) acc += w[d * 4 + k] * xn[base + (long)ls * DD + d];
  }
  emb[idx] += acc;
}

// ---------------------------------------------------------------- selective scan (wave per chain, shuffle scan)
__global__ __launch_bounds__(256) void scan_kernel(const float* __restrict__ u,
                                                   const float* __restrict__ A_log,
                                                   float* __restrict__ hs) {
  int lane = threadIdx.x & 31;
  int wave = threadIdx.x >> 5;
  int chain = blockIdx.x * 8 + wave;   // B*DS chains
  int ds = chain & (DSS - 1);
  int b  = chain >> 4;
  float a = __expf(-__expf(A_log[ds]));
  const int SEG = LL / 32;             // 128 timesteps per lane
  long base = (long)b * LL * DSS + ds;
  int t0 = lane * SEG;
  // pass 1: local segment with h_in = 0
  float h = 0.f;
  for (int t = 0; t < SEG; ++t) h = a * h + u[base + (long)(t0 + t) * DSS];
  float Aseg = a;
#pragma unroll
  for (int i = 0; i < 7; ++i) Aseg *= Aseg;   // a^128
  float Acur = Aseg, Bcur = h;
  // inclusive scan of affine carries (A,B): cur ∘ prev
#pragma unroll
  for (int d = 1; d < 32; d <<= 1) {
    float Ap = __shfl_up(Acur, d, 32);
    float Bp = __shfl_up(Bcur, d, 32);
    if (lane >= d) { Bcur = Bcur + Acur * Bp; Acur = Acur * Ap; }
  }
  float carry = __shfl_up(Bcur, 1, 32);
  if (lane == 0) carry = 0.f;
  // pass 2: recompute with carry, emit
  h = carry;
  for (int t = 0; t < SEG; ++t) {
    h = a * h + u[base + (long)(t0 + t) * DSS];
    hs[base + (long)(t0 + t) * DSS] = h;
  }
}

// ---------------------------------------------------------------- im2col (3x3 conv, pad 1, NHWC)
__global__ __launch_bounds__(256) void im2col_conv(const float* __restrict__ in, bhalf_t* __restrict__ p,
                                                   int IH, int IW, int Cin, int Ctot, int coff,
                                                   int stride, int OH, int OW) {
  int K = Cin * 9;
  long total = (long)BB * OH * OW * K;
  long e = (long)blockIdx.x * 256 + threadIdx.x;
  if (e >= total) return;
  long r = e / K;
  int k = e % K;
  int cin = k / 9, t = k % 9;
  int ky = t / 3, kx = t % 3;
  int b = r / (OH * OW);
  int pp = r % (OH * OW);
  int oy = pp / OW, ox = pp % OW;
  int iy = oy * stride - 1 + ky;
  int ix = ox * stride - 1 + kx;
  float v = 0.f;
  if (iy >= 0 && iy < IH && ix >= 0 && ix < IW)
    v = in[(((long)b * IH + iy) * IW + ix) * Ctot + coff + cin];
  p[e] = (bhalf_t)v;
}

// ---------------------------------------------------------------- im2col for ConvT(k4,s2,p1) parity class
__global__ __launch_bounds__(256) void im2col_convT(const float* __restrict__ in, bhalf_t* __restrict__ p,
                                                    int IH, int IW, int Ctot, int dy, int dx,
                                                    int OHc, int OWc) {
  int K = Ctot * 4;
  long total = (long)BB * OHc * OWc * K;
  long e = (long)blockIdx.x * 256 + threadIdx.x;
  if (e >= total) return;
  long r = e / K;
  int k = e % K;
  int cin = k >> 2, t = k & 3;
  int ty = t >> 1, tx = t & 1;
  int b = r / (OHc * OWc);
  int pp = r % (OHc * OWc);
  int m = pp / OWc, j = pp % OWc;
  int iy = m + ty - 1 + dy;
  int ix = j + tx - 1 + dx;
  float v = 0.f;
  if (iy >= 0 && iy < IH && ix >= 0 && ix < IW)
    v = in[(((long)b * IH + iy) * IW + ix) * Ctot + cin];
  p[e] = (bhalf_t)v;
}

// ---------------------------------------------------------------- ConvT weight repack: Wc[cls][cout][cin*4+t]
__global__ __launch_bounds__(256) void repack_convT(const float* __restrict__ w, float* __restrict__ Wc,
                                                    int Cin, int Cout) {
  long total = (long)4 * Cout * Cin * 4;
  long e = (long)blockIdx.x * 256 + threadIdx.x;
  if (e >= total) return;
  int cls = e / (Cout * Cin * 4);
  int rem = e % (Cout * Cin * 4);
  int cout = rem / (Cin * 4);
  int kk = rem % (Cin * 4);
  int cin = kk >> 2, t = kk & 3;
  int ty = t >> 1, tx = t & 1;
  int dy = cls >> 1, dx = cls & 1;
  int ky = 3 - 2 * ty - dy;
  int kx = 3 - 2 * tx - dx;
  Wc[e] = w[(((long)cin * Cout + cout) * 4 + ky) * 4 + kx];
}

// ---------------------------------------------------------------- Dm transpose [DS,D] -> [D,DS]
__global__ __launch_bounds__(256) void repack_dm(const float* __restrict__ dm, float* __restrict__ o) {
  int e = blockIdx.x * 256 + threadIdx.x;
  if (e >= DD * DSS) return;
  int n = e / DSS, k = e % DSS;
  o[e] = dm[k * DD + n];
}

// ---------------------------------------------------------------- copy emb -> u2 channels [128..256), float4
__global__ __launch_bounds__(256) void copy_sp(const float* __restrict__ emb, float* __restrict__ u2) {
  long e = (long)blockIdx.x * 256 + threadIdx.x;        // vec4 index
  if (e >= (long)BL * (DD / 4)) return;
  long r = e / (DD / 4);
  int c = e % (DD / 4);
  ((float4*)(u2 + r * 256 + 128))[c] = ((const float4*)emb)[e];
}

// ---------------------------------------------------------------- final 1x1 conv (NHWC -> NCHW out), float4
__global__ __launch_bounds__(256) void final_kernel(const float* __restrict__ u2,
                                                    const float* __restrict__ fw,
                                                    const float* __restrict__ fb,
                                                    float* __restrict__ out) {
  long idx = (long)blockIdx.x * 256 + threadIdx.x;
  if (idx >= (long)BB * LL) return;
  int b = idx >> 12;
  int p = idx & (LL - 1);
  float a0 = fb[0], a1 = fb[1];
  const float4* row = (const float4*)(u2 + idx * 256);
  const float4* w0  = (const float4*)fw;
  const float4* w1  = (const float4*)(fw + 256);
  for (int k = 0; k < 64; ++k) {
    float4 v = row[k], c0 = w0[k], c1 = w1[k];
    a0 += v.x * c0.x + v.y * c0.y + v.z * c0.z + v.w * c0.w;
    a1 += v.x * c1.x + v.y * c1.y + v.z * c1.z + v.w * c1.w;
  }
  out[((long)(b * 2 + 0) << 12) + p] = a0;
  out[((long)(b * 2 + 1) << 12) + p] = a1;
}

// ---------------------------------------------------------------- host orchestration
static inline GemmArgs plain(const float* Wt, const float* bias, float* out, const float* res,
                             const float* mul, int M, int N, int K, int lda, int ldc, int coloff, int act) {
  GemmArgs g{};
  g.Wt = Wt; g.bias = bias; g.out = out; g.res = res; g.mul = mul;
  g.M = M; g.N = N; g.K = K; g.lda = lda; g.ldc = ldc; g.col_off = coloff; g.act = act;
  g.map_mode = 0; g.rpi = 1; g.wc = 1; g.img_str = 0; g.rsy = 0; g.rsx = 1; g.roff = 0;
  return g;
}

extern "C" void kernel_launch(void* const* d_in, const int* in_sizes, int n_in,
                              void* d_out, int out_size, void* d_ws, size_t ws_size,
                              hipStream_t stream) {
  (void)in_sizes; (void)n_in; (void)out_size; (void)ws_size;
  const float* x        = (const float*)d_in[0];
  const float* emb_w    = (const float*)d_in[1];
  const float* emb_b    = (const float*)d_in[2];
  const float* pos      = (const float*)d_in[3];
  const float* timee    = (const float*)d_in[4];
  const float* ln1_w    = (const float*)d_in[5];
  const float* ln1_b    = (const float*)d_in[6];
  const float* conv_w   = (const float*)d_in[7];
  const float* conv_b   = (const float*)d_in[8];
  const float* sln_w    = (const float*)d_in[9];
  const float* sln_b    = (const float*)d_in[10];
  const float* A_log    = (const float*)d_in[11];
  const float* blkD     = (const float*)d_in[12];
  const float* in_w     = (const float*)d_in[13];
  const float* in_b     = (const float*)d_in[14];
  const float* x_w      = (const float*)d_in[15];
  const float* x_b      = (const float*)d_in[16];
  const float* out_w    = (const float*)d_in[17];
  const float* out_b    = (const float*)d_in[18];
  const float* ln2_w    = (const float*)d_in[19];
  const float* ln2_b    = (const float*)d_in[20];
  const float* mlp_w1   = (const float*)d_in[21];
  const float* mlp_b1   = (const float*)d_in[22];
  const float* mlp_w2   = (const float*)d_in[23];
  const float* mlp_b2   = (const float*)d_in[24];
  const float* down1_w  = (const float*)d_in[25];
  const float* down1_b  = (const float*)d_in[26];
  const float* down2_w  = (const float*)d_in[27];
  const float* down2_b  = (const float*)d_in[28];
  const float* bott_w   = (const float*)d_in[29];
  const float* bott_b   = (const float*)d_in[30];
  const float* up1_w    = (const float*)d_in[31];
  const float* up1_b    = (const float*)d_in[32];
  const float* up2_w    = (const float*)d_in[33];
  const float* up2_b    = (const float*)d_in[34];
  const float* final_w  = (const float*)d_in[35];
  const float* final_b  = (const float*)d_in[36];

  // workspace carve-up
  char* ws = (char*)d_ws;
  size_t off = 0;
  auto take = [&](size_t bytes) { char* p = ws + off; off = (off + bytes + 255) & ~(size_t)255; return p; };
  float*   emb     = (float*)take((size_t)BL * DD * 4);
  float*   xn      = (float*)take((size_t)BL * DD * 4);
  float*   hbuf    = (float*)take((size_t)BL * 256 * 4);   // xp [BL,128] then mlp hidden [BL,256]
  float*   zbuf    = (float*)take((size_t)BL * DD * 4);
  float*   ubuf    = (float*)take((size_t)BL * DSS * 4);
  float*   hsbuf   = (float*)take((size_t)BL * DSS * 4);
  float*   ybuf    = (float*)take((size_t)BL * DD * 4);
  float*   DmT     = (float*)take((size_t)DD * DSS * 4);
  bhalf_t* patches = (bhalf_t*)take((size_t)16384 * 2304 * 2);
  float*   u1      = (float*)take((size_t)BB * 32 * 32 * 256 * 4);
  float*   x2      = (float*)take((size_t)BB * 16 * 16 * 256 * 4);
  float*   x3      = (float*)take((size_t)BB * 16 * 16 * 256 * 4);
  float*   u2      = (float*)take((size_t)BB * 64 * 64 * 256 * 4);
  float*   Wc      = (float*)take((size_t)4 * 128 * 1024 * 4);

  const dim3 T(256);

  // ---- embed (conv3x3 + pos + time, weighted frame sum)
  embed_kernel<<<BB * HH, T, 0, stream>>>(x, emb_w, emb_b, pos, timee, emb);

  // ---- Mamba blocks
  for (int i = 0; i < 2; ++i) {
    ln_kernel<<<BL / 8, T, 0, stream>>>(emb, ln1_w + i * DD, ln1_b + i * DD, xn, BL);
    dwconv_kernel<<<(BL * DD) / 256, T, 0, stream>>>(xn, conv_w + i * DD * 4, conv_b + i * DD, emb);

    ln_kernel<<<BL / 8, T, 0, stream>>>(emb, sln_w + i * DD, sln_b + i * DD, xn, BL);
    // xp = silu(xn @ in_w[:128].T + in_b[:128]) ; z = sigmoid(rows 128..256)
    gemm_wmma<float><<<dim3(2, BL / 64), T, 0, stream>>>(xn,
        plain(in_w + (size_t)i * 256 * DD, in_b + i * 256, hbuf, nullptr, nullptr,
              BL, 128, 128, 128, 128, 0, /*silu*/3));
    gemm_wmma<float><<<dim3(2, BL / 64), T, 0, stream>>>(xn,
        plain(in_w + (size_t)i * 256 * DD + 128 * DD, in_b + i * 256 + 128, zbuf, nullptr, nullptr,
              BL, 128, 128, 128, 128, 0, /*sigmoid*/4));
    // u = xp @ x_w.T + x_b
    gemm_wmma<float><<<dim3(1, BL / 64), T, 0, stream>>>(hbuf,
        plain(x_w + (size_t)i * DSS * DD, x_b + i * DSS, ubuf, nullptr, nullptr,
              BL, DSS, 128, 128, DSS, 0, 0));
    // selective scan
    scan_kernel<<<(BB * DSS) / 8, T, 0, stream>>>(ubuf, A_log + i * DSS, hsbuf);
    // y = (hs @ Dm) * z
    repack_dm<<<(DD * DSS + 255) / 256, T, 0, stream>>>(blkD + (size_t)i * DSS * DD, DmT);
    gemm_wmma<float><<<dim3(2, BL / 64), T, 0, stream>>>(hsbuf,
        plain(DmT, nullptr, ybuf, nullptr, zbuf, BL, 128, DSS, DSS, 128, 0, 0));
    // emb += y @ out_w.T + out_b
    gemm_wmma<float><<<dim3(2, BL / 64), T, 0, stream>>>(ybuf,
        plain(out_w + (size_t)i * DD * DD, out_b + i * DD, emb, emb, nullptr,
              BL, 128, 128, 128, 128, 0, 0));
    // MLP
    ln_kernel<<<BL / 8, T, 0, stream>>>(emb, ln2_w + i * DD, ln2_b + i * DD, xn, BL);
    gemm_wmma<float><<<dim3(4, BL / 64), T, 0, stream>>>(xn,
        plain(mlp_w1 + (size_t)i * 256 * DD, mlp_b1 + i * 256, hbuf, nullptr, nullptr,
              BL, 256, 128, 128, 256, 0, /*gelu*/2));
    gemm_wmma<float><<<dim3(2, BL / 64), T, 0, stream>>>(hbuf,
        plain(mlp_w2 + (size_t)i * DD * 256, mlp_b2 + i * DD, emb, emb, nullptr,
              BL, 128, 256, 256, 128, 0, 0));
  }

  // ---- UNet (NHWC, implicit GEMM)
  // down1: 64x64x128 -> 32x32x128 (into u1 cols [128..256) == skip x1)
  {
    long tot = (long)BB * 32 * 32 * 128 * 9;
    im2col_conv<<<(tot + 255) / 256, T, 0, stream>>>(emb, patches, 64, 64, 128, 128, 0, 2, 32, 32);
    gemm_wmma<bhalf_t><<<dim3(2, 16384 / 64), T, 0, stream>>>(patches,
        plain(down1_w, down1_b, u1, nullptr, nullptr, 16384, 128, 1152, 1152, 256, 128, 1));
  }
  // down2: 32x32x128 -> 16x16x256
  {
    long tot = (long)BB * 16 * 16 * 128 * 9;
    im2col_conv<<<(tot + 255) / 256, T, 0, stream>>>(u1, patches, 32, 32, 128, 256, 128, 2, 16, 16);
    gemm_wmma<bhalf_t><<<dim3(4, 4096 / 64), T, 0, stream>>>(patches,
        plain(down2_w, down2_b, x2, nullptr, nullptr, 4096, 256, 1152, 1152, 256, 0, 1));
  }
  // bottleneck: 16x16x256 -> 16x16x256
  {
    long tot = (long)BB * 16 * 16 * 256 * 9;
    im2col_conv<<<(tot + 255) / 256, T, 0, stream>>>(x2, patches, 16, 16, 256, 256, 0, 1, 16, 16);
    gemm_wmma<bhalf_t><<<dim3(4, 4096 / 64), T, 0, stream>>>(patches,
        plain(bott_w, bott_b, x3, nullptr, nullptr, 4096, 256, 2304, 2304, 256, 0, 1));
  }
  // up1: ConvT 16->32, 256->128, into u1 cols [0..128)
  repack_convT<<<(4 * 128 * 1024 + 255) / 256, T, 0, stream>>>(up1_w, Wc, 256, 128);
  for (int cls = 0; cls < 4; ++cls) {
    int dy = cls >> 1, dx = cls & 1;
    long tot = (long)BB * 16 * 16 * 256 * 4;
    im2col_convT<<<(tot + 255) / 256, T, 0, stream>>>(x3, patches, 16, 16, 256, dy, dx, 16, 16);
    GemmArgs g = plain(Wc + (size_t)cls * 128 * 1024, up1_b, u1, nullptr, nullptr,
                       4096, 128, 1024, 1024, 256, 0, 1);
    g.map_mode = 1; g.rpi = 256; g.wc = 16; g.img_str = 1024; g.rsy = 64; g.rsx = 2; g.roff = dy * 32 + dx;
    gemm_wmma<bhalf_t><<<dim3(2, 4096 / 64), T, 0, stream>>>(patches, g);
  }
  // up2: ConvT 32->64, 256->128, into u2 cols [0..128)
  repack_convT<<<(4 * 128 * 1024 + 255) / 256, T, 0, stream>>>(up2_w, Wc, 256, 128);
  for (int cls = 0; cls < 4; ++cls) {
    int dy = cls >> 1, dx = cls & 1;
    long tot = (long)BB * 32 * 32 * 256 * 4;
    im2col_convT<<<(tot + 255) / 256, T, 0, stream>>>(u1, patches, 32, 32, 256, dy, dx, 32, 32);
    GemmArgs g = plain(Wc + (size_t)cls * 128 * 1024, up2_b, u2, nullptr, nullptr,
                       16384, 128, 1024, 1024, 256, 0, 1);
    g.map_mode = 1; g.rpi = 1024; g.wc = 32; g.img_str = 4096; g.rsy = 128; g.rsx = 2; g.roff = dy * 64 + dx;
    gemm_wmma<bhalf_t><<<dim3(2, 16384 / 64), T, 0, stream>>>(patches, g);
  }
  // concat skip: sp (= emb) -> u2 cols [128..256)
  copy_sp<<<(BL * (DD / 4)) / 256, T, 0, stream>>>(emb, u2);
  // final 1x1 conv 256->2, NCHW output
  final_kernel<<<(BB * LL) / 256, T, 0, stream>>>(u2, final_w, final_b, (float*)d_out);
}
